// StreamingDurationHead_85109071937712
// MI455X (gfx1250) — matrix operand order repaired and checked by reading, not compiled, not run on hardware
//
#include <hip/hip_runtime.h>
#include <hip/hip_bf16.h>

// ---------------------------------------------------------------------------
// StreamingDurationHead for MI455X (gfx1250, wave32, WMMA).
// All GEMMs: v_wmma_f32_16x16x32_bf16, A operands pre-converted to bf16 so the
// inner loop is pure b128-load + WMMA. Each wave owns a 16x64 strip (4 WMMAs
// per A fragment). Epilogues are template-specialized (no runtime branches).
// ---------------------------------------------------------------------------

#define BB   16
#define UU   2048
#define NR   (BB * UU)     // 32768 rows
#define DD   512
#define MSL  64            // role slots
#define VV   2048          // vocab
#define USH  11            // log2(UU)

typedef __attribute__((ext_vector_type(16))) __bf16          v16bf;
typedef __attribute__((ext_vector_type(16))) unsigned short  u16x16;
typedef __attribute__((ext_vector_type(8)))  float           v8f;

__device__ __forceinline__ unsigned short f2bf(float x) {
  unsigned int u = __float_as_uint(x);
  u += 0x7FFFu + ((u >> 16) & 1u);        // round-to-nearest-even
  return (unsigned short)(u >> 16);
}
__device__ __forceinline__ float gelu_exact(float x) {
  return 0.5f * x * (1.0f + erff(x * 0.70710678118654752440f));
}
__device__ __forceinline__ float wred_sum(float s) {
  #pragma unroll
  for (int off = 16; off > 0; off >>= 1) s += __shfl_xor(s, off, 32);
  return s;
}
__device__ __forceinline__ float wred_max(float s) {
  #pragma unroll
  for (int off = 16; off > 0; off >>= 1) s = fmaxf(s, __shfl_xor(s, off, 32));
  return s;
}

// ---------------------------------------------------------------------------
// EMA scan + feature scalars (serial per batch; 16 batches -> 16 threads)
// ---------------------------------------------------------------------------
__global__ void ema_kernel(const float* __restrict__ la, const float* __restrict__ um,
                           const float* __restrict__ sm, const float* __restrict__ sil,
                           const float* __restrict__ init,
                           float* __restrict__ maskc, float* __restrict__ silc,
                           float* __restrict__ centered, float* __restrict__ silf) {
  int b = threadIdx.x;
  if (b >= BB) return;
  float st = init[b];
  for (int u = 0; u < UU; ++u) {
    int n = b * UU + u;
    float m  = fminf(fmaxf(um[n],  0.f), 1.f);
    float se = fminf(fmaxf(sm[n],  0.f), 1.f);
    float si = fminf(fmaxf(sil[n], 0.f), 1.f);
    float speech = m * se * (1.f - si);
    float rate = st;                                   // state BEFORE unit u
    if (speech > 0.5f) st = 0.95f * st + 0.05f * la[n];
    maskc[n] = m;
    silc[n]  = si;
    centered[n] = (la[n] - rate) * m;
    silf[n] = si * m;
  }
}

// ---------------------------------------------------------------------------
// fp32 -> bf16 converters (direct and transposed into Wt[Ncols][K])
// ---------------------------------------------------------------------------
__global__ void conv_bf16_direct(const float* __restrict__ src,
                                 unsigned short* __restrict__ dst, int n) {
  int i = blockIdx.x * blockDim.x + threadIdx.x;
  if (i < n) dst[i] = f2bf(src[i]);
}
// src is [K x Ncols] row-major (ld = Ncols); dst is [Ncols x K]
__global__ void conv_bf16_T(const float* __restrict__ src,
                            unsigned short* __restrict__ dst, int K, int Ncols) {
  int i = blockIdx.x * blockDim.x + threadIdx.x;
  if (i >= K * Ncols) return;
  int n = i / K, k = i % K;
  dst[i] = f2bf(src[k * Ncols + n]);
}

// ---------------------------------------------------------------------------
// WMMA GEMM: C[NR x Ncols] = Abf[NR x K](bf16) @ Wt^T (bf16) + epilogue(MODE)
// One wave = one 16x64 strip (4 accumulators sharing the A fragment).
// Grid sized exactly -> EXEC all-ones at every WMMA (ISA requirement).
// MODE 0: C = acc                               (score)
// MODE 1: C = (acc + bias_col + s0*v0 + s1*v1 + s2*v2) * maskrow   (in_proj)
// MODE 2: C = acc + bias_col + resid            (conv pointwise, residual)
// MODE 3: C = gelu(acc + bias_bcol[b])          (coarse hidden)
// MODE 4: C = gelu(acc + bias_bcol[b] + s0*v0)  (resid hidden)
// ---------------------------------------------------------------------------
struct Epi {
  const float* bias_col;   // [Ncols]
  const float* bias_bcol;  // [BB x Ncols], indexed (row>>USH)*Ncols + col
  const float* s0; const float* v0;   // rank-1 terms: + s[row]*v[col]
  const float* s1; const float* v1;
  const float* s2; const float* v2;
  const float* resid;      // += resid[row*Ncols+col]
  const float* maskrow;    // *= maskrow[row]
};

template <int MODE>
__global__ void gemm_wmma(const unsigned short* __restrict__ Abf,
                          const int* __restrict__ gather_idx,
                          const unsigned short* __restrict__ gather_base,
                          const unsigned short* __restrict__ Wt,
                          int K, int Ncols, float* __restrict__ C,
                          Epi e, int nstrips) {
  const int wid  = (blockIdx.x * blockDim.x + threadIdx.x) >> 5;
  const int lane = threadIdx.x & 31;
  const int ns = wid % nstrips;          // strip of 4 N-tiles (64 cols)
  const int mt = wid / nstrips;
  const int half = lane >> 4;            // 0: low half, 1: high half
  const int l16  = lane & 15;

  const int arow = mt * 16 + l16;
  const unsigned short* Arow = (MODE == 1)
      ? (gather_base + (size_t)gather_idx[arow] * DD + half * 16)
      : (Abf + (size_t)arow * K + half * 16);
  const int ncol0 = ns * 64;
  // B fragment: lane holds col = tile_col + l16, K {0..15} (half 0) / {16..31}
  const unsigned short* Bbase = Wt + (size_t)(ncol0 + l16) * K + half * 16;

  v8f acc0 = {}, acc1 = {}, acc2 = {}, acc3 = {};
  for (int kk = 0; kk < K; kk += 32) {
    // A fragment: K offsets {0..15} + half*16 (two b128 loads)
    union { u16x16 v; uint4 q[2]; } ua;
    ua.q[0] = *(const uint4*)(Arow + kk);
    ua.q[1] = *(const uint4*)(Arow + kk + 8);
    v16bf a = __builtin_bit_cast(v16bf, ua.v);

    v16bf b0 = __builtin_bit_cast(v16bf, *(const u16x16*)(Bbase + kk));
    v16bf b1 = __builtin_bit_cast(v16bf, *(const u16x16*)(Bbase + (size_t)16 * K + kk));
    v16bf b2 = __builtin_bit_cast(v16bf, *(const u16x16*)(Bbase + (size_t)32 * K + kk));
    v16bf b3 = __builtin_bit_cast(v16bf, *(const u16x16*)(Bbase + (size_t)48 * K + kk));

    acc0 = __builtin_amdgcn_wmma_f32_16x16x32_bf16(false, a, false, b0, (short)0, acc0, false, false);
    acc1 = __builtin_amdgcn_wmma_f32_16x16x32_bf16(false, a, false, b1, (short)0, acc1, false, false);
    acc2 = __builtin_amdgcn_wmma_f32_16x16x32_bf16(false, a, false, b2, (short)0, acc2, false, false);
    acc3 = __builtin_amdgcn_wmma_f32_16x16x32_bf16(false, a, false, b3, (short)0, acc3, false, false);
  }

  // C layout: VGPR r -> row = mt*16 + r + half*8, col = tile_col + l16
  #pragma unroll
  for (int r = 0; r < 8; ++r) {
    const int row = mt * 16 + r + half * 8;
    float sc0 = 0.f, sc1 = 0.f, sc2 = 0.f, mk = 1.f;
    if (MODE == 1) { sc0 = e.s0[row]; sc1 = e.s1[row]; sc2 = e.s2[row]; mk = e.maskrow[row]; }
    if (MODE == 4) { sc0 = e.s0[row]; }
    const int bofs = (MODE >= 3) ? ((row >> USH) * Ncols) : 0;
    float vals[4] = {acc0[r], acc1[r], acc2[r], acc3[r]};
    #pragma unroll
    for (int t = 0; t < 4; ++t) {
      const int col = ncol0 + t * 16 + l16;
      float val = vals[t];
      if (MODE == 1) val = (val + e.bias_col[col] + sc0 * e.v0[col]
                                 + sc1 * e.v1[col] + sc2 * e.v2[col]) * mk;
      if (MODE == 2) val = val + e.bias_col[col] + e.resid[(size_t)row * Ncols + col];
      if (MODE == 3) val = gelu_exact(val + e.bias_bcol[bofs + col]);
      if (MODE == 4) val = gelu_exact(val + e.bias_bcol[bofs + col] + sc0 * e.v0[col]);
      C[(size_t)row * Ncols + col] = val;
    }
  }
}

// ---------------------------------------------------------------------------
// Causal dilated depthwise conv (K=3) + gelu -> bf16 (feeds pointwise GEMM)
// ---------------------------------------------------------------------------
__global__ void dwconv_gelu(const float* __restrict__ ht, unsigned short* __restrict__ outp,
                            const float* __restrict__ dww, const float* __restrict__ dwb,
                            int dil) {
  int i = blockIdx.x * blockDim.x + threadIdx.x;     // over NR*DD
  int d = i & (DD - 1);
  int u = (i >> 9) & (UU - 1);
  float w0 = dww[d * 3 + 0], w1 = dww[d * 3 + 1], w2 = dww[d * 3 + 2];
  float x0 = (u >= 2 * dil) ? ht[i - 2 * dil * DD] : 0.f;
  float x1 = (u >= dil)     ? ht[i - dil * DD]     : 0.f;
  float x2 = ht[i];
  outp[i] = f2bf(gelu_exact(w0 * x0 + w1 * x1 + w2 * x2 + dwb[d]));
}

// ---------------------------------------------------------------------------
// LayerNorm over D=512 (one wave per row) * mask -> bf16 query
// ---------------------------------------------------------------------------
__global__ void ln_kernel(const float* __restrict__ x, const float* __restrict__ g,
                          const float* __restrict__ bb, const float* __restrict__ maskrow,
                          unsigned short* __restrict__ q_out) {
  int wid  = (blockIdx.x * blockDim.x + threadIdx.x) >> 5;
  int lane = threadIdx.x & 31;
  const float* row = x + (size_t)wid * DD;
  float v[16];
  const float4* p = (const float4*)(row + lane * 16);
  #pragma unroll
  for (int q = 0; q < 4; ++q) {
    float4 f = p[q];
    v[q*4+0]=f.x; v[q*4+1]=f.y; v[q*4+2]=f.z; v[q*4+3]=f.w;
  }
  float s = 0.f;
  #pragma unroll
  for (int j = 0; j < 16; ++j) s += v[j];
  float mu = wred_sum(s) * (1.f / DD);
  float vs = 0.f;
  #pragma unroll
  for (int j = 0; j < 16; ++j) { float d = v[j] - mu; vs += d * d; }
  float inv = rsqrtf(wred_sum(vs) * (1.f / DD) + 1e-5f);
  float mk = maskrow[wid];
  unsigned short* qrow = q_out + (size_t)wid * DD;
  #pragma unroll
  for (int j = 0; j < 16; ++j) {
    int d = lane * 16 + j;
    qrow[d] = f2bf(((v[j] - mu) * inv * g[d] + bb[d]) * mk);
  }
}

// ---------------------------------------------------------------------------
// Role-slot softmax + weighted reductions (one wave per row; 64 slots)
// ---------------------------------------------------------------------------
__global__ void attn_kernel(const float* __restrict__ score, const float* __restrict__ cov,
                            const float* __restrict__ rval, const float* __restrict__ rvar,
                            const float* __restrict__ maskc,
                            float* __restrict__ rv_out, float* __restrict__ rconf_out) {
  int wid  = (blockIdx.x * blockDim.x + threadIdx.x) >> 5;
  int lane = threadIdx.x & 31;
  int b = wid >> USH;
  float mk = maskc[wid];
  int m0 = lane, m1 = lane + 32;
  float s0 = score[(size_t)wid * MSL + m0] + logf(fmaxf(cov[b * MSL + m0], 1e-4f));
  float s1 = score[(size_t)wid * MSL + m1] + logf(fmaxf(cov[b * MSL + m1], 1e-4f));
  if (mk <= 0.f) { s0 = -1e4f; s1 = -1e4f; }
  float mx = wred_max(fmaxf(s0, s1));
  float e0 = expf(s0 - mx), e1 = expf(s1 - mx);
  float den = wred_sum(e0 + e1);
  float a0 = e0 / den * mk, a1 = e1 / den * mk;
  float rv = wred_sum(a0 * rval[b * MSL + m0] + a1 * rval[b * MSL + m1]);
  float va = wred_sum(a0 * rvar[b * MSL + m0] + a1 * rvar[b * MSL + m1]);
  float rc = wred_sum(a0 * cov [b * MSL + m0] + a1 * cov [b * MSL + m1]);
  if (lane == 0) {
    float rvarc = fmaxf(va, 1e-4f);
    float rcovc = fmaxf(rc, 1e-4f);
    rv_out[wid]    = rv;
    rconf_out[wid] = rcovc / (rcovc + rvarc);
  }
}

// ---------------------------------------------------------------------------
// spk projection + folded per-batch head biases (tiny)
// ---------------------------------------------------------------------------
__global__ void spk_kernel(const float* __restrict__ spk, const float* __restrict__ w,
                           const float* __restrict__ bias, float* __restrict__ out) {
  int b = blockIdx.x, d = threadIdx.x;
  float s = bias[d];
  const float* sp = spk + b * DD;
  for (int k = 0; k < DD; ++k) s += sp[k] * w[k * DD + d];
  out[b * DD + d] = s;
}

__global__ void head_bias_kernel(const float* __restrict__ summ, const float* __restrict__ spkp,
                                 const float* __restrict__ cw1, const float* __restrict__ cb1,
                                 const float* __restrict__ rw1, const float* __restrict__ rb1,
                                 float* __restrict__ biasC, float* __restrict__ biasR) {
  int b = blockIdx.x, d = threadIdx.x;
  float sc = cb1[d], sr = rb1[d];
  for (int k = 0; k < DD; ++k) {
    float su = summ[b * DD + k], sp = spkp[b * DD + k];
    sc += su * cw1[(DD + k) * DD + d] + sp * cw1[(2 * DD + k) * DD + d];
    sr += su * rw1[(DD + k) * DD + d] + sp * rw1[(2 * DD + k) * DD + d];
  }
  biasC[b * DD + d] = sc;
  biasR[b * DD + d] = sr;
}

// coarse[row] = tanh(dot(X[row,:], w) + bias) * scale  (one wave per row)
__global__ void rowdot_tanh(const float* __restrict__ X, const float* __restrict__ w,
                            const float* __restrict__ bias, float scale,
                            float* __restrict__ out) {
  int wid  = (blockIdx.x * blockDim.x + threadIdx.x) >> 5;
  int lane = threadIdx.x & 31;
  const float* row = X + (size_t)wid * DD;
  float s = 0.f;
  #pragma unroll
  for (int j = 0; j < 16; ++j) { int d = lane * 16 + j; s += row[d] * w[d]; }
  s = wred_sum(s);
  if (lane == 0) out[wid] = tanhf(s + bias[0]) * scale;
}

// resid row-dot + confidence fusion + clip + mask -> d_out
__global__ void final_kernel(const float* __restrict__ X, const float* __restrict__ rw2,
                             const float* __restrict__ rb2, const float* __restrict__ coarse,
                             const float* __restrict__ rv, const float* __restrict__ rconf,
                             const float* __restrict__ silc, const float* __restrict__ maskc,
                             float* __restrict__ out) {
  int wid  = (blockIdx.x * blockDim.x + threadIdx.x) >> 5;
  int lane = threadIdx.x & 31;
  const float* row = X + (size_t)wid * DD;
  float s = 0.f;
  #pragma unroll
  for (int j = 0; j < 16; ++j) { int d = lane * 16 + j; s += row[d] * rw2[d]; }
  s = wred_sum(s);
  if (lane == 0) {
    float resid = tanhf(s + rb2[0]) * 0.35f;
    float raw = coarse[wid] + resid;
    float cf = rconf[wid];
    float fused = (1.f - cf) * raw + cf * rv[wid];
    float cap = (silc[wid] > 0.5f) ? 0.35f : 1.2f;
    out[wid] = fminf(fmaxf(fused, -cap), cap) * maskc[wid];
  }
}

// ---------------------------------------------------------------------------
extern "C" void kernel_launch(void* const* d_in, const int* in_sizes, int n_in,
                              void* d_out, int out_size, void* d_ws, size_t ws_size,
                              hipStream_t stream) {
  const int*   content_units = (const int*)  d_in[0];
  const float* log_anchor    = (const float*)d_in[1];
  const float* unit_mask     = (const float*)d_in[2];
  const float* sealed_mask   = (const float*)d_in[3];
  const float* silence_mask  = (const float*)d_in[4];
  const float* local_rate    = (const float*)d_in[5];
  const float* summary_state = (const float*)d_in[6];
  const float* spk_embed     = (const float*)d_in[7];
  const float* role_value    = (const float*)d_in[8];
  const float* role_var      = (const float*)d_in[9];
  const float* role_cov      = (const float*)d_in[10];
  const float* unit_emb      = (const float*)d_in[11];
  const float* in_proj_w     = (const float*)d_in[12];
  const float* in_proj_b     = (const float*)d_in[13];
  const float* dw_w          = (const float*)d_in[14];
  const float* dw_b          = (const float*)d_in[15];
  const float* pw_w          = (const float*)d_in[16];
  const float* pw_b          = (const float*)d_in[17];
  const float* ln_g          = (const float*)d_in[18];
  const float* ln_b          = (const float*)d_in[19];
  const float* role_key      = (const float*)d_in[20];
  const float* spk_w         = (const float*)d_in[21];
  const float* spk_b         = (const float*)d_in[22];
  const float* c_w1          = (const float*)d_in[23];
  const float* c_b1          = (const float*)d_in[24];
  const float* c_w2          = (const float*)d_in[25];
  const float* c_b2          = (const float*)d_in[26];
  const float* r_w1          = (const float*)d_in[27];
  const float* r_b1          = (const float*)d_in[28];
  const float* r_w2          = (const float*)d_in[29];
  const float* r_b2          = (const float*)d_in[30];
  float* out = (float*)d_out;

  char* ws = (char*)d_ws;
  float*          ht    = (float*)(ws);                                  //  64 MB f32
  float*          tmpf  = (float*)(ws + ((size_t)64  << 20));            //  64 MB f32 (head phase)
  unsigned short* tmpbf = (unsigned short*)(ws + ((size_t)64 << 20));    //  same region as bf16 (conv phase)
  unsigned short* qbf   = (unsigned short*)(ws + ((size_t)128 << 20));   //  32 MB bf16 query
  float*          score = (float*)(ws + ((size_t)160 << 20));            //   8 MB
  unsigned short* embbf = (unsigned short*)(ws + ((size_t)168 << 20));   //   2 MB bf16 embedding
  unsigned short* wt    = (unsigned short*)(ws + ((size_t)170 << 20));   // 512 KB bf16 weights
  float* sb       = (float*)(ws + ((size_t)171 << 20));
  float* maskc    = sb + 0 * NR;
  float* silc     = sb + 1 * NR;
  float* centered = sb + 2 * NR;
  float* silf     = sb + 3 * NR;
  float* coarse   = sb + 4 * NR;
  float* rv       = sb + 5 * NR;
  float* rconf    = sb + 6 * NR;
  float* biasC    = sb + 7 * NR;
  float* biasR    = biasC + BB * DD;
  float* spkp     = biasR + BB * DD;

  // grid: each wave owns 16 rows x 64 cols; 8 waves / block, exact division
  const int blocks_D = (NR / 16) * (DD  / 64) / 8;   // 2048
  const int blocks_M = (NR / 16) * (MSL / 64) / 8;   //  256

  // 1) EMA scan + scalar features; bf16 embedding table
  ema_kernel<<<1, BB, 0, stream>>>(log_anchor, unit_mask, sealed_mask, silence_mask,
                                   local_rate, maskc, silc, centered, silf);
  conv_bf16_direct<<<(VV * DD) / 256, 256, 0, stream>>>(unit_emb, embbf, VV * DD);

  // 2) in_proj: ht = (gather(emb) @ W[0:512] + la*W[512] + cent*W[513] + silf*W[514] + b) * mask
  conv_bf16_T<<<(DD * DD) / 256, 256, 0, stream>>>(in_proj_w, wt, DD, DD);
  {
    Epi e = {};
    e.bias_col = in_proj_b;
    e.s0 = log_anchor; e.v0 = in_proj_w + (size_t)512 * DD;
    e.s1 = centered;   e.v1 = in_proj_w + (size_t)513 * DD;
    e.s2 = silf;       e.v2 = in_proj_w + (size_t)514 * DD;
    e.maskrow = maskc;
    gemm_wmma<1><<<blocks_D, 256, 0, stream>>>(nullptr, content_units, embbf,
                                               wt, DD, DD, ht, e, DD / 64);
  }

  // 3) four dilated conv blocks: tmpbf = bf16(gelu(dwconv(ht))); ht += tmpbf @ pw_w[i]^T + pw_b[i]
  const int dils[4] = {1, 2, 4, 8};
  for (int i = 0; i < 4; ++i) {
    dwconv_gelu<<<(NR * DD) / 256, 256, 0, stream>>>(ht, tmpbf,
        dw_w + (size_t)i * DD * 3, dw_b + (size_t)i * DD, dils[i]);
    // einsum 'bcu,dc->bdu' => Wt[Ncols=d][K=c] == pw_w[i] directly
    conv_bf16_direct<<<(DD * DD) / 256, 256, 0, stream>>>(pw_w + (size_t)i * DD * DD,
                                                          wt, DD * DD);
    Epi e = {};
    e.bias_col = pw_b + (size_t)i * DD;
    e.resid = ht;
    gemm_wmma<2><<<blocks_D, 256, 0, stream>>>(tmpbf, nullptr, nullptr,
                                               wt, DD, DD, ht, e, DD / 64);
  }

  // 4) LayerNorm * mask -> bf16 query
  ln_kernel<<<NR / 8, 256, 0, stream>>>(ht, ln_g, ln_b, maskc, qbf);

  // 5) role-slot attention: score = query @ role_key^T  (Wt == role_key)
  conv_bf16_direct<<<(MSL * DD) / 256, 256, 0, stream>>>(role_key, wt, MSL * DD);
  {
    Epi e = {};
    gemm_wmma<0><<<blocks_M, 256, 0, stream>>>(qbf, nullptr, nullptr,
                                               wt, DD, MSL, score, e, MSL / 64);
  }
  attn_kernel<<<NR / 8, 256, 0, stream>>>(score, role_cov, role_value, role_var,
                                          maskc, rv, rconf);

  // 6) per-batch folded head biases
  spk_kernel<<<BB, DD, 0, stream>>>(spk_embed, spk_w, spk_b, spkp);
  head_bias_kernel<<<BB, DD, 0, stream>>>(summary_state, spkp, c_w1, c_b1,
                                          r_w1, r_b1, biasC, biasR);

  // 7) coarse head: tmpf = gelu(query @ c_w1[0:512] + biasC[b]); coarse = tanh(tmpf@c_w2+b)*0.2
  conv_bf16_T<<<(DD * DD) / 256, 256, 0, stream>>>(c_w1, wt, DD, DD);
  {
    Epi e = {};
    e.bias_bcol = biasC;
    gemm_wmma<3><<<blocks_D, 256, 0, stream>>>(qbf, nullptr, nullptr,
                                               wt, DD, DD, tmpf, e, DD / 64);
  }
  rowdot_tanh<<<NR / 8, 256, 0, stream>>>(tmpf, c_w2, c_b2, 0.2f, coarse);

  // 8) resid head: tmpf = gelu(query @ r_w1[0:512] + biasR[b] + coarse*r_w1[1536])
  conv_bf16_T<<<(DD * DD) / 256, 256, 0, stream>>>(r_w1, wt, DD, DD);
  {
    Epi e = {};
    e.bias_bcol = biasR;
    e.s0 = coarse; e.v0 = r_w1 + (size_t)(3 * DD) * DD;
    gemm_wmma<4><<<blocks_D, 256, 0, stream>>>(qbf, nullptr, nullptr,
                                               wt, DD, DD, tmpf, e, DD / 64);
  }

  // 9) fuse + clip + mask -> out
  final_kernel<<<NR / 8, 256, 0, stream>>>(tmpf, r_w2, r_b2, coarse, rv, rconf,
                                           silc, maskc, out);
}
